// GRU1d_2456721293686
// MI455X (gfx1250) — compile-verified, matrix operand order checked by reading, not run on hardware
//
#include <hip/hip_runtime.h>
#include <stdint.h>

#define B_   16
#define T_   1024
#define H_   1024
#define IN_  256
#define G3_  3072
#define YEL  (B_ * T_ * H_)   /* 16,777,216 y elements in d_out */

typedef __attribute__((ext_vector_type(16))) __bf16 v16bf;
typedef __attribute__((ext_vector_type(8)))  float  v8f;
typedef uint32_t u32x4 __attribute__((ext_vector_type(4)));
typedef uint32_t u32x8 __attribute__((ext_vector_type(8)));

static __device__ __forceinline__ uint16_t f2bf(float f) {
  union { float f; uint32_t u; } c; c.f = f;
  uint32_t u = c.u + 0x7FFFu + ((c.u >> 16) & 1u);   // round-to-nearest-even
  return (uint16_t)(u >> 16);
}
static __device__ __forceinline__ float bf2f(uint16_t h) {
  union { uint32_t u; float f; } c; c.u = ((uint32_t)h) << 16;
  return c.f;
}
// element i (0..7) of a packed 8 x bf16 quad-dword, as f32
static __device__ __forceinline__ float bfx(u32x4 q, int i) {
  uint32_t w = q[i >> 1];
  return bf2f((uint16_t)((i & 1) ? (w >> 16) : (w & 0xFFFFu)));
}

// 16-bit WMMA operand fragment per ISA layout: this lane's row, 8 bf16 at p[0..7]
// (VGPR0-3) and 8 bf16 at p[16..23] (VGPR4-7). Caller applies the lane-half K shift.
static __device__ __forceinline__ v16bf ldfrag(const uint16_t* p) {
  u32x4 lo = *(const u32x4*)p;
  u32x4 hi = *(const u32x4*)(p + 16);
  u32x8 u  = __builtin_shufflevector(lo, hi, 0, 1, 2, 3, 4, 5, 6, 7);
  v16bf v;
  __builtin_memcpy(&v, &u, sizeof(v));
  return v;
}

__global__ void cvt_f32_bf16(const float* __restrict__ src,
                             uint16_t* __restrict__ dst, int n) {
  for (int i = blockIdx.x * blockDim.x + threadIdx.x; i < n;
       i += gridDim.x * blockDim.x)
    dst[i] = f2bf(src[i]);
}

// gx GEMM: A[M=B*T, K] (bf16 row-major) x W[3072, K]^T (bf16 row-major) + bias.
// Output stored SWIZZLED for the scan: Cs[((t*3072)+g)*16 + b]  (bf16),
// matching the WMMA C fragment (lane = gate column g, VGPR = batch b).
__global__ void gemm_gx(const uint16_t* __restrict__ A,
                        const uint16_t* __restrict__ W,
                        const float*    __restrict__ bias,
                        uint16_t*       __restrict__ Cs,
                        int M, int K) {
  const int lane = threadIdx.x & 31;
  const int wave = threadIdx.x >> 5;
  const int wpb  = blockDim.x >> 5;
  const int l15  = lane & 15;
  const int half = lane >> 4;
  const int nTN  = G3_ / 16;
  const int nT   = (M / 16) * nTN;

  for (int tile = blockIdx.x * wpb + wave; tile < nT; tile += gridDim.x * wpb) {
    const int m0 = (tile / nTN) * 16;                  // 16 consecutive rows = same b
    const int n0 = (tile % nTN) * 16;
    const uint16_t* arow = A + (size_t)(m0 + l15) * K;
    const uint16_t* wrow = W + (size_t)(n0 + l15) * K;
    v8f acc = {};
    for (int k = 0; k < K; k += 32) {
      const int klo = k + half * 8;
      v16bf af = ldfrag(arow + klo);
      v16bf bf = ldfrag(wrow + klo);
      acc = __builtin_amdgcn_wmma_f32_16x16x32_bf16(false, af, false, bf,
                                                    (short)0, acc, false, false);
    }
    const float bn = bias[n0 + l15];
#pragma unroll
    for (int v = 0; v < 8; ++v) {
      const int r = m0 + v + half * 8;                 // row = b*T + t
      const int b = r >> 10;                           // / T_
      const int t = r & (T_ - 1);
      Cs[((size_t)t * G3_ + n0 + l15) * 16 + b] = f2bf(acc[v] + bn);
    }
  }
}

// Persistent GRU scan for one layer. 16 blocks x 128 threads = 64 waves;
// wave g owns hidden columns [16g, 16g+16). Per step: coalesced gx fragment
// loads (overlapped), stage h into LDS, 3 WMMA strip-GEMMs (r,z,n) over
// K=1024 streaming w_hh from L2, register-resident h_prev gate math, grid barrier.
__global__ void gru_scan(const uint16_t* __restrict__ gxs,   // [T,3072,16] bf16 swizzled
                         const uint16_t* __restrict__ whh,   // [3072, 1024] bf16
                         const float*    __restrict__ bhh,   // [3072] f32
                         uint16_t* __restrict__ hbuf0,       // [B*H] bf16 ping
                         uint16_t* __restrict__ hbuf1,       // [B*H] bf16 pong
                         uint16_t* __restrict__ y_bf,        // layer0 y (bf16) or null
                         float*    __restrict__ y_f32,       // layer1 y (f32) or null
                         float*    __restrict__ h_last,      // [B*H] f32
                         unsigned* __restrict__ barrier_cnt) {
  __shared__ uint16_t hsm[B_ * H_];                    // 32 KB of 320 KB/WGP
  const int lane = threadIdx.x & 31;
  const int wave = threadIdx.x >> 5;
  const int gw   = blockIdx.x * (blockDim.x >> 5) + wave;  // 0..63
  const int col0 = gw * 16;
  const int l15  = lane & 15;
  const int half = lane >> 4;

  const uint16_t* wr = whh + (size_t)(col0 + l15) * H_;
  const uint16_t* wz = whh + (size_t)(H_ + col0 + l15) * H_;
  const uint16_t* wn = whh + (size_t)(2 * H_ + col0 + l15) * H_;
  const float br  = bhh[col0 + l15];
  const float bz  = bhh[H_ + col0 + l15];
  const float bnn = bhh[2 * H_ + col0 + l15];

  // h_prev held in registers across the whole scan: hp[v] = h[b = v+half*8][col0+l15]
  float hp[8];
#pragma unroll
  for (int v = 0; v < 8; ++v)
    hp[v] = bf2f(hbuf0[(size_t)(v + half * 8) * H_ + col0 + l15]);

  unsigned epoch = 0;
  for (int t = 0; t < T_; ++t) {
    const uint16_t* hcur  = (t & 1) ? hbuf1 : hbuf0;
    uint16_t*       hnext = (t & 1) ? hbuf0 : hbuf1;

    // Coalesced gx fragments for this step (one b128 per gate), issued early
    // so their latency hides under the staging + WMMA work below.
    const size_t gxt = (size_t)t * G3_ * 16 + (size_t)half * 8;
    const u32x4 gqr = *(const u32x4*)(gxs + gxt + (size_t)(col0 + l15) * 16);
    const u32x4 gqz = *(const u32x4*)(gxs + gxt + (size_t)(H_ + col0 + l15) * 16);
    const u32x4 gqn = *(const u32x4*)(gxs + gxt + (size_t)(2 * H_ + col0 + l15) * 16);

    // Stage h (16x1024 bf16) into LDS, whole workgroup.
    {
      const u32x4* s = (const u32x4*)hcur;
      u32x4*       d = (u32x4*)hsm;
      for (int i = threadIdx.x; i < (B_ * H_) / 8; i += (int)blockDim.x) d[i] = s[i];
    }
    __syncthreads();

    // Three 16x16 strips of gh = h @ w_hh^T, K = 1024. A fragment shared.
    v8f ar = {}, az = {}, an = {};
    for (int k = 0; k < H_; k += 32) {
      const int klo = k + half * 8;
      v16bf af = ldfrag(hsm + (size_t)l15 * H_ + klo);    // A from LDS
      v16bf b0 = ldfrag(wr + klo);
      ar = __builtin_amdgcn_wmma_f32_16x16x32_bf16(false, af, false, b0,
                                                   (short)0, ar, false, false);
      v16bf b1 = ldfrag(wz + klo);
      az = __builtin_amdgcn_wmma_f32_16x16x32_bf16(false, af, false, b1,
                                                   (short)0, az, false, false);
      v16bf b2 = ldfrag(wn + klo);
      an = __builtin_amdgcn_wmma_f32_16x16x32_bf16(false, af, false, b2,
                                                   (short)0, an, false, false);
    }

    // Gate math, fully in registers. C layout: lane = column, VGPR v = batch.
#pragma unroll
    for (int v = 0; v < 8; ++v) {
      const int m = v + half * 8;
      const float r  = 1.0f / (1.0f + __expf(-(bfx(gqr, v) + ar[v] + br)));
      const float z  = 1.0f / (1.0f + __expf(-(bfx(gqz, v) + az[v] + bz)));
      const float n  = tanhf(bfx(gqn, v) + r * (an[v] + bnn));
      const float hn = (1.0f - z) * n + z * hp[v];
      hp[v] = hn;
      hnext[(size_t)m * H_ + col0 + l15] = f2bf(hn);
      if (y_bf)  y_bf[((size_t)m * T_ + t) * H_ + col0 + l15] = f2bf(hn);
      if (y_f32) y_f32[((size_t)m * T_ + t) * H_ + col0 + l15] = hn;
    }

    // Grid-wide barrier: all 16 blocks must publish hnext before next step.
    __syncthreads();
    if (threadIdx.x == 0) {
      __threadfence();
      atomicAdd(barrier_cnt, 1u);
      ++epoch;
      const unsigned target = (unsigned)gridDim.x * epoch;
      while (atomicAdd(barrier_cnt, 0u) < target) __builtin_amdgcn_s_sleep(2);
    }
    __syncthreads();
  }

  // Final hidden state from the register copy.
#pragma unroll
  for (int v = 0; v < 8; ++v)
    h_last[(size_t)(v + half * 8) * H_ + col0 + l15] = hp[v];
}

extern "C" void kernel_launch(void* const* d_in, const int* in_sizes, int n_in,
                              void* d_out, int out_size, void* d_ws, size_t ws_size,
                              hipStream_t stream) {
  (void)in_sizes; (void)n_in; (void)out_size; (void)ws_size;
  const float* x    = (const float*)d_in[0];
  const float* h0   = (const float*)d_in[1];
  const float* wih0 = (const float*)d_in[2];
  const float* whh0 = (const float*)d_in[3];
  const float* bih0 = (const float*)d_in[4];
  const float* bhh0 = (const float*)d_in[5];
  const float* wih1 = (const float*)d_in[6];
  const float* whh1 = (const float*)d_in[7];
  const float* bih1 = (const float*)d_in[8];
  const float* bhh1 = (const float*)d_in[9];
  float* out = (float*)d_out;

  char* ws = (char*)d_ws;
  size_t off = 0;
  auto take = [&](size_t bytes) -> void* {
    void* p = (void*)(ws + off);
    off += (bytes + 255) & ~(size_t)255;
    return p;
  };
  unsigned* cnt   = (unsigned*)take(2 * sizeof(unsigned));
  uint16_t* xbf   = (uint16_t*)take((size_t)B_ * T_ * IN_ * 2);
  uint16_t* wih0b = (uint16_t*)take((size_t)G3_ * IN_ * 2);
  uint16_t* whh0b = (uint16_t*)take((size_t)G3_ * H_ * 2);
  uint16_t* wih1b = (uint16_t*)take((size_t)G3_ * H_ * 2);
  uint16_t* whh1b = (uint16_t*)take((size_t)G3_ * H_ * 2);
  uint16_t* gxb   = (uint16_t*)take((size_t)B_ * T_ * G3_ * 2);  // swizzled, reused
  uint16_t* y0b   = (uint16_t*)take((size_t)B_ * T_ * H_ * 2);
  uint16_t* hb0   = (uint16_t*)take((size_t)B_ * H_ * 2);
  uint16_t* hb1   = (uint16_t*)take((size_t)B_ * H_ * 2);
  uint16_t* hb2   = (uint16_t*)take((size_t)B_ * H_ * 2);
  uint16_t* hb3   = (uint16_t*)take((size_t)B_ * H_ * 2);

  hipMemsetAsync(cnt, 0, 2 * sizeof(unsigned), stream);   // deterministic barrier state

  const dim3 cb(256);
  cvt_f32_bf16<<<2048, cb, 0, stream>>>(x,    xbf,   B_ * T_ * IN_);
  cvt_f32_bf16<<<512,  cb, 0, stream>>>(wih0, wih0b, G3_ * IN_);
  cvt_f32_bf16<<<1024, cb, 0, stream>>>(whh0, whh0b, G3_ * H_);
  cvt_f32_bf16<<<1024, cb, 0, stream>>>(wih1, wih1b, G3_ * H_);
  cvt_f32_bf16<<<1024, cb, 0, stream>>>(whh1, whh1b, G3_ * H_);
  cvt_f32_bf16<<<64,   cb, 0, stream>>>(h0,            hb0, B_ * H_);
  cvt_f32_bf16<<<64,   cb, 0, stream>>>(h0 + B_ * H_,  hb2, B_ * H_);

  // Layer 0: input-gate GEMM (K=256), then sequential scan.
  gemm_gx<<<1024, 256, 0, stream>>>(xbf, wih0b, bih0, gxb, B_ * T_, IN_);
  gru_scan<<<16, 128, 0, stream>>>(gxb, whh0b, bhh0, hb0, hb1,
                                   y0b, nullptr, out + YEL, cnt + 0);
  // Layer 1: input-gate GEMM over layer-0 output (K=1024), then scan into d_out.
  gemm_gx<<<1024, 256, 0, stream>>>(y0b, wih1b, bih1, gxb, B_ * T_, H_);
  gru_scan<<<16, 128, 0, stream>>>(gxb, whh1b, bhh1, hb2, hb3,
                                   nullptr, out, out + YEL + B_ * H_, cnt + 1);
}
